// LSA_69063074119916
// MI455X (gfx1250) — matrix operand order repaired and checked by reading, not compiled, not run on hardware
//
#include <hip/hip_runtime.h>
#include <hip/hip_bf16.h>

typedef __attribute__((ext_vector_type(16))) _Float16 v16h;
typedef __attribute__((ext_vector_type(8)))  float    v8f;

#define L_ELEMS 256   // H*W = 16*16
#define NBLOCKS 512   // B*N = 8*64
#define THREADS 256   // 8 waves (wave32)

#define LOG2E 1.4426950408889634f

__global__ __launch_bounds__(THREADS) void lsa_rank1_softmax_wmma(
    const float* __restrict__ patch,
    const float* __restrict__ wk, const float* __restrict__ bk,
    const float* __restrict__ wq, const float* __restrict__ bq,
    const float* __restrict__ wv, const float* __restrict__ bv,
    float* __restrict__ out)
{
    __shared__ float     s_k[L_ELEMS];
    __shared__ float     s_q[L_ELEMS];
    __shared__ _Float16  s_vh[L_ELEMS];   // v pre-converted to f16 for the B matrix
    __shared__ float     s_red[16];
    __shared__ float     s_qmax, s_qmin;

    const int tid  = threadIdx.x;
    const int bn   = blockIdx.x;          // flattened (b, n)
    const int wave = tid >> 5;
    const int lane = tid & 31;

    // ---- stage k, q, v in LDS (one element per thread) ----
    const float x  = patch[bn * L_ELEMS + tid];
    const float kf = wk[0] * x + bk[0];
    const float qf = wq[0] * x + bq[0];
    const float vf = wv[0] * x + bv[0];
    s_k[tid]  = kf;
    s_q[tid]  = qf;
    s_vh[tid] = (_Float16)vf;

    // ---- qmax / qmin reduction (rank-1 scores => free softmax stabilizer) ----
    float qmax = qf, qmin = qf;
    #pragma unroll
    for (int off = 16; off > 0; off >>= 1) {
        qmax = fmaxf(qmax, __shfl_xor(qmax, off, 32));
        qmin = fminf(qmin, __shfl_xor(qmin, off, 32));
    }
    if (lane == 0) { s_red[wave] = qmax; s_red[8 + wave] = qmin; }
    __syncthreads();
    if (tid == 0) {
        float mx = s_red[0], mn = s_red[8];
        #pragma unroll
        for (int w = 1; w < 8; ++w) {
            mx = fmaxf(mx, s_red[w]);
            mn = fminf(mn, s_red[8 + w]);
        }
        s_qmax = mx; s_qmin = mn;
    }
    __syncthreads();
    qmax = s_qmax;
    qmin = s_qmin;

    // ---- per-wave: two 16-row tiles; WMMA fuses sum(exp*v) and sum(exp) ----
    const int M    = lane & 15;   // A row / D column index
    const int half = lane >> 4;   // K-half selector per ISA 16-bit layouts

    const _Float16 one_h = (_Float16)1.0f;

    for (int t = wave * 2; t < wave * 2 + 2; ++t) {
        const int   row = t * 16 + M;
        // Fold log2(e) into k once per row: exp(ki*q - mi) == exp2(ki2*q - mi2)
        const float ki2 = s_k[row] * LOG2E;
        const float mi2 = (ki2 >= 0.0f) ? (ki2 * qmax) : (ki2 * qmin);

        v8f acc = {};   // f32 accumulator: N=0 -> numerator, N=1 -> denominator
        #pragma unroll
        for (int jc = 0; jc < 8; ++jc) {
            const int jb = jc * 32;

            // A (16x32 f16): lane (M, half); VGPR p<4 -> K = 8*half + 2p{,+1};
            // p>=4 -> K = 16 + 8*half + 2(p-4){,+1}.
            // E[i,j] = exp2(ki2*q_j - mi2) in (0,1].
            // __builtin_amdgcn_exp2f = bare v_exp_f32 (no subnormal fixup):
            // args are <= 0, and anything below 2^-24 dies in f16 anyway.
            v16h a;
            #pragma unroll
            for (int p = 0; p < 8; ++p) {
                const int k0 = (p < 4) ? (half * 8 + 2 * p)
                                       : (16 + half * 8 + 2 * (p - 4));
                const float e0 = __builtin_amdgcn_exp2f(ki2 * s_q[jb + k0]     - mi2);
                const float e1 = __builtin_amdgcn_exp2f(ki2 * s_q[jb + k0 + 1] - mi2);
                a[2 * p]     = (_Float16)e0;
                a[2 * p + 1] = (_Float16)e1;
            }

            // B (32x16 f16): lane holds column N = lane&15; element s -> same K as A.
            // Column 0 = v_j, column 1 = 1.0 (row sum), rest = 0.
            v16h bm = {};
            #pragma unroll
            for (int s = 0; s < 16; ++s) {
                const int kk = (s < 8) ? (half * 8 + s) : (16 + half * 8 + (s - 8));
                bm[s] = (M == 0) ? s_vh[jb + kk] : ((M == 1) ? one_h : (_Float16)0.0f);
            }

            // D = A x B + C  (v_wmma_f32_16x16x32_f16)
            acc = __builtin_amdgcn_wmma_f32_16x16x32_f16(
                /*neg_a=*/false, a, /*neg_b=*/false, bm,
                /*c_mod=*/(short)0, acc, /*reuse_a=*/false, /*reuse_b=*/false);
        }

        // D layout: lane N = lane&15; VGPR r holds row (r + 8*half).
        // numerator at N==0 lanes, denominator at N==1 lanes -> pair via shfl_xor(1).
        #pragma unroll
        for (int r = 0; r < 8; ++r) {
            const float den = __shfl_xor(acc[r], 1, 32);
            if (M == 0) {
                // denominator >= 1 (max element contributes exp2(0)=1): rcp is safe
                out[bn * L_ELEMS + t * 16 + r + 8 * half] =
                    acc[r] * __builtin_amdgcn_rcpf(den);
            }
        }
    }
}

extern "C" void kernel_launch(void* const* d_in, const int* in_sizes, int n_in,
                              void* d_out, int out_size, void* d_ws, size_t ws_size,
                              hipStream_t stream) {
    const float* patch = (const float*)d_in[0];
    const float* wk    = (const float*)d_in[1];
    const float* bk    = (const float*)d_in[2];
    const float* wq    = (const float*)d_in[3];
    const float* bq    = (const float*)d_in[4];
    const float* wv    = (const float*)d_in[5];
    const float* bv    = (const float*)d_in[6];
    float* out = (float*)d_out;

    lsa_rank1_softmax_wmma<<<NBLOCKS, THREADS, 0, stream>>>(
        patch, wk, bk, wq, bq, wv, bv, out);
}
